// RelativeAttention_26216480374954
// MI455X (gfx1250) — matrix operand order repaired and checked by reading, not compiled, not run on hardware
//
#include <hip/hip_runtime.h>

typedef __attribute__((ext_vector_type(16))) _Float16 v16h;
typedef __attribute__((ext_vector_type(8)))  float    v8f;

#define DIM   512
#define HEADS 8
#define SEQ   2048
#define BATCH 4
#define HD    64

union V16HU { v16h v; unsigned int u[8]; };

__device__ inline unsigned int pack_h2(float a, float b) {
    union { _Float16 h[2]; unsigned int u; } t;
    t.h[0] = (_Float16)a; t.h[1] = (_Float16)b;
    return t.u;
}

// A-fragment (16-bit A 16x32, ISA 7.12.2): lane half hh owns K runs [hh*8, hh*8+8) and [16+hh*8,...)
__device__ inline void load_a16(const _Float16* p, int hh, V16HU& f) {
    uint4 a = *(const uint4*)(p + (hh << 3));
    uint4 b = *(const uint4*)(p + 16 + (hh << 3));
    f.u[0] = a.x; f.u[1] = a.y; f.u[2] = a.z; f.u[3] = a.w;
    f.u[4] = b.x; f.u[5] = b.y; f.u[6] = b.z; f.u[7] = b.w;
}
// B-fragment (16-bit B 32x16): lane half hh owns K runs [hh*8, hh*8+8) and [16+hh*8,...)
__device__ inline void load_b16(const _Float16* p, int hh, V16HU& f) {
    uint4 a = *(const uint4*)(p + (hh << 3));
    uint4 b = *(const uint4*)(p + 16 + (hh << 3));
    f.u[0] = a.x; f.u[1] = a.y; f.u[2] = a.z; f.u[3] = a.w;
    f.u[4] = b.x; f.u[5] = b.y; f.u[6] = b.z; f.u[7] = b.w;
}
// Same fragments from f32 memory, converting to f16 in registers.
__device__ inline void load_a16_f32(const float* p, int hh, V16HU& f) {
    const float* p0 = p + (hh << 3);
    float4 a0 = ((const float4*)p0)[0];
    float4 a1 = ((const float4*)p0)[1];
    float4 a2 = *(const float4*)(p0 + 16);
    float4 a3 = *(const float4*)(p0 + 20);
    f.u[0] = pack_h2(a0.x, a0.y); f.u[1] = pack_h2(a0.z, a0.w);
    f.u[2] = pack_h2(a1.x, a1.y); f.u[3] = pack_h2(a1.z, a1.w);
    f.u[4] = pack_h2(a2.x, a2.y); f.u[5] = pack_h2(a2.z, a2.w);
    f.u[6] = pack_h2(a3.x, a3.y); f.u[7] = pack_h2(a3.z, a3.w);
}
__device__ inline void load_b16_f32(const float* p, int hh, V16HU& f) {
    const float* p0 = p + (hh << 3);
    float4 b0 = ((const float4*)p0)[0], b1 = ((const float4*)p0)[1];
    float4 b2 = *(const float4*)(p0 + 16), b3 = *(const float4*)(p0 + 20);
    f.u[0] = pack_h2(b0.x, b0.y); f.u[1] = pack_h2(b0.z, b0.w);
    f.u[2] = pack_h2(b1.x, b1.y); f.u[3] = pack_h2(b1.z, b1.w);
    f.u[4] = pack_h2(b2.x, b2.y); f.u[5] = pack_h2(b2.z, b2.w);
    f.u[6] = pack_h2(b3.x, b3.y); f.u[7] = pack_h2(b3.z, b3.w);
}

// ---------------------------------------------------------------------------
// Kernel 1: qkv = x @ Wqkv.T  -> q[b][h][n][d], k[b][h][n][d], vT[b][h][d][n]
// ---------------------------------------------------------------------------
__global__ __launch_bounds__(128)
void qkv_gemm_kernel(const float* __restrict__ x, const float* __restrict__ Wqkv,
                     _Float16* __restrict__ qs, _Float16* __restrict__ ks,
                     _Float16* __restrict__ vt)
{
    const int lane = threadIdx.x & 31;
    const int wave = threadIdx.x >> 5;
    // wave-uniform tile id, forced into an SGPR so all derived bases are scalar
    const int tile = __builtin_amdgcn_readfirstlane(blockIdx.x * 4 + wave);
    const int tile_m = tile / 96;               // 49152 tiles = 512 x 96
    const int tile_n = tile % 96;
    const int r  = lane & 15;
    const int hh = lane >> 4;
    const int row0 = tile_m * 16;
    const int col0 = tile_n * 16;

    const float* arow = x    + (size_t)(row0 + r) * DIM;
    const float* brow = Wqkv + (size_t)(col0 + r) * DIM;

    v8f c = {};
    for (int k0 = 0; k0 < DIM; k0 += 32) {
        V16HU av, bv;
        load_a16_f32(arow + k0, hh, av);
        load_b16_f32(brow + k0, hh, bv);
        c = __builtin_amdgcn_wmma_f32_16x16x32_f16(false, av.v, false, bv.v,
                                                   (short)0, c, false, false);
    }

    // Branch-free scatter: scalar-selected base pointer + scalar strides.
    // q/k: idx = (bh*SEQ + n)*HD + d   (stride_n=HD, stride_d=1)
    // v^T: idx = (bh*HD + d)*SEQ + n   (stride_n=1,  stride_d=SEQ)
    const int which = __builtin_amdgcn_readfirstlane(col0 >> 9);     // 0=q 1=k 2=v
    const int h     = __builtin_amdgcn_readfirstlane((col0 >> 6) & 7);
    const int bb    = __builtin_amdgcn_readfirstlane(row0 >> 11);
    const int nbase = row0 & (SEQ - 1);
    const int d     = (col0 & 63) + r;          // col0 multiple of 16: no carry
    _Float16* mat  = (which == 0) ? qs : (which == 1) ? ks : vt;
    const int sn   = (which == 2) ? 1 : HD;     // scalar
    const int sd   = (which == 2) ? SEQ : 1;    // scalar
    _Float16* dst  = mat + (size_t)(bb * HEADS + h) * (SEQ * HD)
                         + (size_t)d * sd + (size_t)(nbase + (hh << 3)) * sn;
    #pragma unroll
    for (int v = 0; v < 8; ++v)
        dst[(size_t)v * sn] = (_Float16)c[v];

// ---------------------------------------------------------------------------
}

// ---------------------------------------------------------------------------
// Kernel 2: flash attention. One wave per (b, h, 16 query rows); 64-key chunks.
// ---------------------------------------------------------------------------
__global__ __launch_bounds__(256)
void flash_attn_kernel(const _Float16* __restrict__ qs, const _Float16* __restrict__ ks,
                       const _Float16* __restrict__ vt, const float* __restrict__ mask,
                       const float* __restrict__ Btab, _Float16* __restrict__ ao)
{
    __shared__ _Float16 plds[8 * 16 * 64];      // per-wave 16x64 P staging (16 KB)
    const int lane = threadIdx.x & 31;
    const int wave = threadIdx.x >> 5;
    const int tile = __builtin_amdgcn_readfirstlane(blockIdx.x * 8 + wave); // 4096 tiles
    const int b  = tile >> 10;
    const int h  = (tile >> 7) & 7;
    const int qt = tile & 127;
    const int n0 = qt * 16;
    const int r  = lane & 15;
    const int hh = lane >> 4;

    _Float16* wl = plds + wave * (16 * 64);

    const size_t bh = (size_t)(b * HEADS + h);
    const _Float16* qbase = qs + (bh * SEQ + n0) * HD;   // SGPR base
    const _Float16* kbase = ks + bh * SEQ * HD;          // SGPR base
    const _Float16* vbase = vt + bh * HD * SEQ;          // SGPR base
    const float*    mbase = mask + ((size_t)b * SEQ + n0) * SEQ;
    const float*    bbase = Btab + h * (2 * SEQ - 1);

    // Q tile 16x64 as two A-fragments (K = 0..31, 32..63)
    V16HU qa0, qa1;
    {
        const _Float16* qrow = qbase + r * HD;
        load_a16(qrow, hh, qa0);
        load_a16(qrow + 32, hh, qa1);
    }

    const float scale = 0.125f;  // 64^-0.5
    float M[8], L[8];
    #pragma unroll
    for (int v = 0; v < 8; ++v) { M[v] = -3.0e38f; L[v] = 0.0f; }
    v8f o0 = {}, o1 = {}, o2 = {}, o3 = {};

    for (int kb = 0; kb < SEQ; kb += 64) {
        if (kb + 64 < SEQ) {                    // uniform branch: prefetch next K chunk
            const _Float16* pf = kbase + (size_t)(kb + 64 + (lane << 1)) * HD;
            __builtin_prefetch(pf, 0, 1);
            __builtin_prefetch(pf + HD, 0, 1);
        }

        float s[4][8];
        #pragma unroll
        for (int sub = 0; sub < 4; ++sub) {
            const int kbb = kb + (sub << 4);
            const _Float16* krow = kbase + (size_t)(kbb + r) * HD;
            V16HU kf0, kf1;
            load_b16(krow, hh, kf0);
            load_b16(krow + 32, hh, kf1);
            v8f sacc = {};
            sacc = __builtin_amdgcn_wmma_f32_16x16x32_f16(false, qa0.v, false, kf0.v,
                                                          (short)0, sacc, false, false);
            sacc = __builtin_amdgcn_wmma_f32_16x16x32_f16(false, qa1.v, false, kf1.v,
                                                          (short)0, sacc, false, false);
            const int kidx = kbb + r;
            #pragma unroll
            for (int v = 0; v < 8; ++v) {
                const int qrow_i = (hh << 3) + v;
                const float bias = bbase[(n0 + qrow_i) - kidx + (SEQ - 1)];
                const float mk   = mbase[(size_t)qrow_i * SEQ + kidx];
                s[sub][v] = sacc[v] * scale + bias + mk;
            }
        }

        // online softmax (row reductions across the 16-lane C-layout half)
        float cf[8];
        #pragma unroll
        for (int v = 0; v < 8; ++v) {
            float cm = fmaxf(fmaxf(s[0][v], s[1][v]), fmaxf(s[2][v], s[3][v]));
            #pragma unroll
            for (int mw = 1; mw < 16; mw <<= 1)
                cm = fmaxf(cm, __shfl_xor(cm, mw, 32));
            const float mn = fmaxf(M[v], cm);
            cf[v] = __expf(M[v] - mn);
            float ps = 0.0f;
            const int m_ = (hh << 3) + v;
            #pragma unroll
            for (int sub = 0; sub < 4; ++sub) {
                const float p = __expf(s[sub][v] - mn);
                ps += p;
                wl[m_ * 64 + (sub << 4) + r] = (_Float16)p;  // C-layout -> A-layout via LDS
            }
            #pragma unroll
            for (int mw = 1; mw < 16; mw <<= 1)
                ps += __shfl_xor(ps, mw, 32);
            L[v] = L[v] * cf[v] + ps;
            M[v] = mn;
        }
        #pragma unroll
        for (int v = 0; v < 8; ++v) {
            o0[v] *= cf[v]; o1[v] *= cf[v]; o2[v] *= cf[v]; o3[v] *= cf[v];
        }

        // P as two A-fragments (keys kb..kb+31, kb+32..kb+63); same-wave LDS is in-order
        V16HU pa0, pa1;
        load_a16(wl + r * 64, hh, pa0);
        load_a16(wl + r * 64 + 32, hh, pa1);

        // O += P @ V  (V^T layout: key dim contiguous)
        #pragma unroll
        for (int t = 0; t < 4; ++t) {
            const _Float16* vrow = vbase + (size_t)((t << 4) + r) * SEQ + kb;
            V16HU vb0, vb1;
            load_b16(vrow, hh, vb0);
            load_b16(vrow + 32, hh, vb1);
            v8f& ot = (t == 0) ? o0 : (t == 1) ? o1 : (t == 2) ? o2 : o3;
            ot = __builtin_amdgcn_wmma_f32_16x16x32_f16(false, pa0.v, false, vb0.v,
                                                        (short)0, ot, false, false);
            ot = __builtin_amdgcn_wmma_f32_16x16x32_f16(false, pa1.v, false, vb1.v,
                                                        (short)0, ot, false, false);
        }
    }

    #pragma unroll
    for (int v = 0; v < 8; ++v) {
        const float inv = 1.0f / L[v];
        const int qidx = n0 + (hh << 3) + v;
        _Float16* orow = ao + ((size_t)b * SEQ + qidx) * DIM + h * HD;
        orow[ 0 + r] = (_Float16)(o0[v] * inv);
        orow[16 + r] = (_Float16)(o1[v] * inv);
        orow[32 + r] = (_Float16)(o2[v] * inv);
        orow[48 + r] = (_Float16)(o3[v] * inv);
    }
}

// ---------------------------------------------------------------------------
// Kernel 3: out = ao @ Wproj.T + bproj   (f32 output)
// ---------------------------------------------------------------------------
__global__ __launch_bounds__(128)
void proj_gemm_kernel(const _Float16* __restrict__ ao, const float* __restrict__ Wproj,
                      const float* __restrict__ bproj, float* __restrict__ out)
{
    const int lane = threadIdx.x & 31;
    const int wave = threadIdx.x >> 5;
    const int tile = __builtin_amdgcn_readfirstlane(blockIdx.x * 4 + wave); // 16384 tiles
    const int tile_m = tile >> 5;
    const int tile_n = tile & 31;
    const int r  = lane & 15;
    const int hh = lane >> 4;
    const int row0 = tile_m * 16;
    const int col0 = tile_n * 16;

    const _Float16* arow = ao    + (size_t)(row0 + r) * DIM;
    const float*    brow = Wproj + (size_t)(col0 + r) * DIM;

    v8f c = {};
    for (int k0 = 0; k0 < DIM; k0 += 32) {
        V16HU av, bv;
        load_a16(arow + k0, hh, av);
        load_b16_f32(brow + k0, hh, bv);
        c = __builtin_amdgcn_wmma_f32_16x16x32_f16(false, av.v, false, bv.v,
                                                   (short)0, c, false, false);
    }
    const float bias = bproj[col0 + r];
    float* drow = out + (size_t)(row0 + (hh << 3)) * DIM + col0 + r;
    #pragma unroll
    for (int v = 0; v < 8; ++v)
        drow[(size_t)v * DIM] = c[v] + bias;
}

// ---------------------------------------------------------------------------
extern "C" void kernel_launch(void* const* d_in, const int* in_sizes, int n_in,
                              void* d_out, int out_size, void* d_ws, size_t ws_size,
                              hipStream_t stream)
{
    const float* x     = (const float*)d_in[0];
    const float* mask  = (const float*)d_in[1];
    const float* Wqkv  = (const float*)d_in[2];
    const float* Btab  = (const float*)d_in[3];
    const float* Wproj = (const float*)d_in[4];
    const float* bproj = (const float*)d_in[5];
    float* out = (float*)d_out;

    const size_t QSZ = (size_t)BATCH * HEADS * SEQ * HD;  // 4.19M elems
    _Float16* qs = (_Float16*)d_ws;                       // 8 MB
    _Float16* ks = qs + QSZ;                              // 8 MB
    _Float16* vt = ks + QSZ;                              // 8 MB (transposed V)
    _Float16* ao = vt + QSZ;                              // 8 MB

    qkv_gemm_kernel <<<12288, 128, 0, stream>>>(x, Wqkv, qs, ks, vt);
    flash_attn_kernel<<<  512, 256, 0, stream>>>(qs, ks, vt, mask, Btab, ao);
    proj_gemm_kernel <<< 4096, 128, 0, stream>>>(ao, Wproj, bproj, out);
}
